// GATLinkPrediction_73787538145484
// MI455X (gfx1250) — matrix operand order repaired and checked by reading, not compiled, not run on hardware
//
#include <hip/hip_runtime.h>

#define HEADS 2

typedef __attribute__((ext_vector_type(16))) _Float16 v16h;
typedef __attribute__((ext_vector_type(8)))  _Float16 v8h;
typedef __attribute__((ext_vector_type(8)))  float    v8f;

// ---------- monotone float<->uint encoding for atomic segment-max ----------
__device__ __forceinline__ unsigned enc_f32(float f) {
  unsigned b = __float_as_uint(f);
  return (b & 0x80000000u) ? ~b : (b | 0x80000000u);
}
__device__ __forceinline__ float dec_f32(unsigned e) {
  unsigned b = (e & 0x80000000u) ? (e & 0x7FFFFFFFu) : ~e;
  return __uint_as_float(b);
}

// ---------- f32 -> f16 conversion (layer input for WMMA) ----------
__global__ void k_f32_to_f16(const float* __restrict__ in, _Float16* __restrict__ out,
                             long long n) {
  long long i = (long long)blockIdx.x * blockDim.x + threadIdx.x;
  if (i < n) out[i] = (_Float16)in[i];
}

// ---------- W[K,M] f32 -> Wt[M,K] f16 (transpose so B-fragments are contiguous) ----------
__global__ void k_w_transpose_f16(const float* __restrict__ W, _Float16* __restrict__ Wt,
                                  int K, int M) {
  int tid = blockIdx.x * blockDim.x + threadIdx.x;
  if (tid >= K * M) return;
  int k = tid / M, m = tid % M;
  Wt[(long long)m * K + k] = (_Float16)W[tid];
}

// ---------- GEMM: C[N,M] = A[N,128] * B[128,M], B given transposed as Bt[M,128] ----------
// One wave per 16x16 output tile, K=128 => 4 x v_wmma_f32_16x16x32_f16.
__global__ void __launch_bounds__(32)
k_gemm_wmma(const _Float16* __restrict__ A, const _Float16* __restrict__ Bt,
            float* __restrict__ C, int M) {
  constexpr int K = 128;
  const int lane = threadIdx.x;
  const int half = lane >> 4;    // 0 or 1
  const int m16  = lane & 15;
  const int colBase = blockIdx.x * 16;
  const long long rowBase = (long long)blockIdx.y * 16;

  const _Float16* arow = A  + (rowBase + m16) * K;              // row = m16 for both halves
  const _Float16* bcol = Bt + (long long)(colBase + m16) * K;   // col = m16 for both halves

  v8f c = {};
#pragma unroll
  for (int kb = 0; kb < K; kb += 32) {
    // A 16x32 f16 fragment: element e <-> K = kb + e + 16*half (matches B)
    v8h a_lo = *(const v8h*)(arow + kb + half * 16);       // K = kb + half*16 + 0..7
    v8h a_hi = *(const v8h*)(arow + kb + half * 16 + 8);   // K = kb + half*16 + 8..15
    v16h a, b;
#pragma unroll
    for (int i = 0; i < 8; ++i) { a[i] = a_lo[i]; a[8 + i] = a_hi[i]; }
    // B 32x16 f16 fragment: element e <-> K = kb + e + 16*half, column = lane&15
    b = *(const v16h*)(bcol + kb + half * 16);
    c = __builtin_amdgcn_wmma_f32_16x16x32_f16(false, a, false, b, (short)0, c,
                                               false, false);
  }
  // C/D layout: vgpr r <-> row r + 8*half, column lane&15
  float* crow = C + (rowBase + 8 * half) * M + colBase + m16;
#pragma unroll
  for (int r = 0; r < 8; ++r) crow[(long long)r * M] = c[r];
}

// ---------- attention logits: s[n,h] = xl[n,h,:].a_src[h,:], d likewise ----------
__global__ void k_sd(const float* __restrict__ xl, const float* __restrict__ asrc,
                     const float* __restrict__ adst, float* __restrict__ sv,
                     float* __restrict__ dv, int n, int Hh, int Cc) {
  int tid = blockIdx.x * blockDim.x + threadIdx.x;
  if (tid >= n * Hh) return;
  int node = tid / Hh, h = tid % Hh;
  const float* xp = xl + (long long)node * Hh * Cc + h * Cc;
  const float* as = asrc + h * Cc;
  const float* ad = adst + h * Cc;
  float ss = 0.f, dd = 0.f;
  for (int c = 0; c < Cc; ++c) { float v = xp[c]; ss += v * as[c]; dd += v * ad[c]; }
  sv[tid] = ss;
  dv[tid] = dd;
}

__device__ __forceinline__ void edge_nodes(const int* ei, int E, int e, int& sn, int& dn) {
  if (e < E) { sn = ei[e]; dn = ei[E + e]; }        // edge_index rows [src..., dst...]
  else       { sn = dn = e - E; }                    // appended self loops
}

// ---------- pass 1: segment max of leaky_relu(s[src]+d[dst]) over dst ----------
__global__ void k_edge_max(const int* __restrict__ ei, int E, int n,
                           const float* __restrict__ sv, const float* __restrict__ dv,
                           unsigned* __restrict__ emax, int Hh) {
  long long tid = (long long)blockIdx.x * blockDim.x + threadIdx.x;
  long long tot = (long long)(E + n) * Hh;
  if (tid >= tot) return;
  int e = (int)(tid / Hh), h = (int)(tid % Hh);
  int sn, dn; edge_nodes(ei, E, e, sn, dn);
  float x = sv[sn * Hh + h] + dv[dn * Hh + h];
  x = (x > 0.f) ? x : 0.2f * x;
  atomicMax(&emax[dn * Hh + h], enc_f32(x));
}

// ---------- pass 2: ex = exp(e - emax[dst]); denom[dst] += ex ----------
__global__ void k_edge_expsum(const int* __restrict__ ei, int E, int n,
                              const float* __restrict__ sv, const float* __restrict__ dv,
                              const unsigned* __restrict__ emax, float* __restrict__ denom,
                              float* __restrict__ exbuf, int Hh) {
  long long tid = (long long)blockIdx.x * blockDim.x + threadIdx.x;
  long long tot = (long long)(E + n) * Hh;
  if (tid >= tot) return;
  int e = (int)(tid / Hh), h = (int)(tid % Hh);
  int sn, dn; edge_nodes(ei, E, e, sn, dn);
  float x = sv[sn * Hh + h] + dv[dn * Hh + h];
  x = (x > 0.f) ? x : 0.2f * x;
  float ex = expf(x - dec_f32(emax[dn * Hh + h]));
  exbuf[tid] = ex;                                   // tid == e*Hh + h
  atomicAdd(&denom[dn * Hh + h], ex);
}

// ---------- pass 3: out[dst] += xl[src] * (ex/denom[dst]) ----------
__global__ void k_edge_agg(const int* __restrict__ ei, int E, int n,
                           const float* __restrict__ xl, const float* __restrict__ exbuf,
                           const float* __restrict__ denom, float* __restrict__ outb,
                           int Hh, int Cc) {
  const int F = Hh * Cc;
  long long tid = (long long)blockIdx.x * blockDim.x + threadIdx.x;
  long long tot = (long long)(E + n) * F;
  if (tid >= tot) return;
  int e = (int)(tid / F);
  int f = (int)(tid % F);
  int h = f / Cc;
  int sn, dn; edge_nodes(ei, E, e, sn, dn);
  float alpha = exbuf[(long long)e * Hh + h] / denom[dn * Hh + h];
  atomicAdd(&outb[(long long)dn * F + f], xl[(long long)sn * F + f] * alpha);
}

// ---------- bias + ELU, emit f16 for next layer's WMMA input ----------
__global__ void k_bias_elu_f16(const float* __restrict__ acc, const float* __restrict__ b,
                               _Float16* __restrict__ out, long long ntot, int M) {
  long long i = (long long)blockIdx.x * blockDim.x + threadIdx.x;
  if (i >= ntot) return;
  float v = acc[i] + b[(int)(i % M)];
  v = (v > 0.f) ? v : expm1f(v);
  out[i] = (_Float16)v;
}

// ---------- final layer: bias add into d_out (f32) ----------
__global__ void k_bias_out(const float* __restrict__ acc, const float* __restrict__ b,
                           float* __restrict__ out, long long ntot, int M) {
  long long i = (long long)blockIdx.x * blockDim.x + threadIdx.x;
  if (i >= ntot) return;
  out[i] = acc[i] + b[(int)(i % M)];
}

extern "C" void kernel_launch(void* const* d_in, const int* in_sizes, int n_in,
                              void* d_out, int out_size, void* d_ws, size_t ws_size,
                              hipStream_t stream) {
  const float* x   = (const float*)d_in[0];
  const int*   ei  = (const int*)d_in[1];
  const float* Ws[3]  = {(const float*)d_in[2], (const float*)d_in[6], (const float*)d_in[10]};
  const float* ass[3] = {(const float*)d_in[3], (const float*)d_in[7], (const float*)d_in[11]};
  const float* ads[3] = {(const float*)d_in[4], (const float*)d_in[8], (const float*)d_in[12]};
  const float* bs[3]  = {(const float*)d_in[5], (const float*)d_in[9], (const float*)d_in[13]};

  const int Dd = 128;
  const int n  = in_sizes[0] / Dd;   // 50000 (multiple of 16)
  const int E  = in_sizes[1] / 2;    // 800000
  const int Et = E + n;
  const int Ms[3] = {128, 128, 64};
  const int Hs[3] = {HEADS, HEADS, 1};

  // ---- workspace carve-up (ws assumed >= ~73 MB, base 256B aligned) ----
  float*    xl    = (float*)d_ws;                       // N*128 f32
  float*    outb  = xl + (size_t)n * 128;               // N*128 f32
  float*    sbuf  = outb + (size_t)n * 128;             // N*H
  float*    dbuf  = sbuf + (size_t)n * HEADS;           // N*H
  unsigned* emax  = (unsigned*)(dbuf + (size_t)n * HEADS);  // N*H
  float*    denom = (float*)(emax + (size_t)n * HEADS);     // N*H
  float*    exbuf = denom + (size_t)n * HEADS;          // (E+N)*H
  _Float16* hf16  = (_Float16*)(exbuf + (size_t)Et * HEADS);// N*128 f16 (16B aligned)
  _Float16* wt16  = hf16 + (size_t)n * 128;             // 128*128 f16 (32B aligned)

  // initial input: x -> f16
  {
    long long tot = (long long)n * 128;
    k_f32_to_f16<<<(unsigned)((tot + 255) / 256), 256, 0, stream>>>(x, hf16, tot);
  }

  for (int L = 0; L < 3; ++L) {
    const int M = Ms[L], Hh = Hs[L], Cc = M / Hh;
    const int K = 128;

    k_w_transpose_f16<<<(K * M + 255) / 256, 256, 0, stream>>>(Ws[L], wt16, K, M);

    dim3 g(M / 16, n / 16);
    k_gemm_wmma<<<g, 32, 0, stream>>>(hf16, wt16, xl, M);

    {
      int tot = n * Hh;
      k_sd<<<(tot + 255) / 256, 256, 0, stream>>>(xl, ass[L], ads[L], sbuf, dbuf, n, Hh, Cc);
    }

    hipMemsetAsync(emax, 0, (size_t)n * Hh * sizeof(unsigned), stream);
    hipMemsetAsync(denom, 0, (size_t)n * Hh * sizeof(float), stream);
    hipMemsetAsync(outb, 0, (size_t)n * M * sizeof(float), stream);

    {
      long long tot = (long long)Et * Hh;
      unsigned nb = (unsigned)((tot + 255) / 256);
      k_edge_max<<<nb, 256, 0, stream>>>(ei, E, n, sbuf, dbuf, emax, Hh);
      k_edge_expsum<<<nb, 256, 0, stream>>>(ei, E, n, sbuf, dbuf, emax, denom, exbuf, Hh);
    }
    {
      long long tot = (long long)Et * M;
      k_edge_agg<<<(unsigned)((tot + 255) / 256), 256, 0, stream>>>(ei, E, n, xl, exbuf,
                                                                    denom, outb, Hh, Cc);
    }

    long long tot = (long long)n * M;
    unsigned nb = (unsigned)((tot + 255) / 256);
    if (L < 2) {
      k_bias_elu_f16<<<nb, 256, 0, stream>>>(outb, bs[L], hf16, tot, M);
    } else {
      k_bias_out<<<nb, 256, 0, stream>>>(outb, bs[L], (float*)d_out, tot, M);
    }
  }
}
